// GraphEncoder_61263413510238
// MI455X (gfx1250) — compile-verified, matrix operand order checked by reading, not run on hardware
//
#include <hip/hip_runtime.h>
#include <hip/hip_bf16.h>

// ---------- types ----------
typedef __bf16 bf16_t;
typedef __attribute__((ext_vector_type(16))) __bf16 bf16x16;
typedef __attribute__((ext_vector_type(8)))  float  f32x8;

// ---------- generic batched WMMA bf16 GEMM ----------
// C[b,h] (MxN) = alpha * A[b,h](MxK, row-major) @ B[b,h]^T (B stored NxK row-major) + bias
// Every operand fragment load is K-contiguous -> global_load_b128 pairs.
// One wave computes a 32x32 tile = 2x2 WMMA tiles (4 x v_wmma_f32_16x16x32_bf16 per K-step).
struct GemmArgs {
    const bf16_t* A; const bf16_t* B; const float* bias;
    float* Cf; bf16_t* Cb;
    int M, N, K, lda, ldb, ldc;
    long long sAb, sAh, sBb, sBh, sCb, sCh;
    int NH, transC, relu;
    float alpha;
};

__device__ inline bf16x16 load_frag(const bf16_t* row, int kb, int lh) {
    union { bf16x16 v; unsigned u[8]; } f;
    const unsigned* p = (const unsigned*)(row + kb);
    #pragma unroll
    for (int j = 0; j < 4; ++j) {           // ISA 7.12.2: VGPR j <- K pair (lh*8+2j),
        f.u[j]     = p[lh * 4 + j];         //             VGPR j+4 <- K pair (16+lh*8+2j)
        f.u[4 + j] = p[8 + lh * 4 + j];
    }
    return f.v;
}

__global__ __launch_bounds__(256)
void k_gemm_bf16(GemmArgs g) {
    const int lane = threadIdx.x;          // wave32
    const int w    = threadIdx.y;          // 8 waves: 2 (M) x 4 (N)
    const int z    = blockIdx.z;
    const int b    = z / g.NH;
    const int hh   = z % g.NH;
    const int m0   = (blockIdx.y * 2 + (w >> 2)) * 32;
    const int n0   = (blockIdx.x * 4 + (w & 3)) * 32;
    if (m0 >= g.M || n0 >= g.N) return;    // wave-uniform; EXEC stays all-ones for WMMA

    const bf16_t* Ap = g.A + (long long)b * g.sAb + (long long)hh * g.sAh;
    const bf16_t* Bp = g.B + (long long)b * g.sBb + (long long)hh * g.sBh;
    const int lr = lane & 15;
    const int lh = lane >> 4;

    // clamp second tiles (duplicate compute, guarded store) to keep loads in range
    const int mA1 = (m0 + 16 < g.M) ? m0 + 16 : m0;
    const int nB1 = (n0 + 16 < g.N) ? n0 + 16 : n0;

    const bf16_t* Arow0 = Ap + (size_t)(m0  + lr) * g.lda;
    const bf16_t* Arow1 = Ap + (size_t)(mA1 + lr) * g.lda;
    const bf16_t* Brow0 = Bp + (size_t)(n0  + lr) * g.ldb;
    const bf16_t* Brow1 = Bp + (size_t)(nB1 + lr) * g.ldb;

    f32x8 acc00 = {0.f,0.f,0.f,0.f,0.f,0.f,0.f,0.f};
    f32x8 acc01 = acc00, acc10 = acc00, acc11 = acc00;

    for (int kb = 0; kb < g.K; kb += 32) {
        // prefetch next K-tile (global_prefetch_b8)
        __builtin_prefetch((const void*)(Arow0 + kb + 32), 0, 1);
        __builtin_prefetch((const void*)(Brow0 + kb + 32), 0, 1);
        bf16x16 a0 = load_frag(Arow0, kb, lh);
        bf16x16 a1 = load_frag(Arow1, kb, lh);
        bf16x16 b0 = load_frag(Brow0, kb, lh);
        bf16x16 b1 = load_frag(Brow1, kb, lh);
        acc00 = __builtin_amdgcn_wmma_f32_16x16x32_bf16(false, a0, false, b0, (short)0, acc00, false, false);
        acc01 = __builtin_amdgcn_wmma_f32_16x16x32_bf16(false, a0, false, b1, (short)0, acc01, false, false);
        acc10 = __builtin_amdgcn_wmma_f32_16x16x32_bf16(false, a1, false, b0, (short)0, acc10, false, false);
        acc11 = __builtin_amdgcn_wmma_f32_16x16x32_bf16(false, a1, false, b1, (short)0, acc11, false, false);
    }

    float*  Cfp = g.Cf ? g.Cf + (long long)b * g.sCb + (long long)hh * g.sCh : nullptr;
    bf16_t* Cbp = g.Cb ? g.Cb + (long long)b * g.sCb + (long long)hh * g.sCh : nullptr;

    f32x8 accs[2][2] = {{acc00, acc01}, {acc10, acc11}};
    #pragma unroll
    for (int tm = 0; tm < 2; ++tm) {
        const int rowb = m0 + 16 * tm;
        if (rowb >= g.M) continue;
        #pragma unroll
        for (int tn = 0; tn < 2; ++tn) {
            const int nb = n0 + 16 * tn;
            if (nb >= g.N) continue;
            const int col = nb + lr;
            const float bv = g.bias ? g.bias[col] : 0.f;
            const f32x8 a = accs[tm][tn];
            if (g.transC) {
                // store C^T: per-lane 8 contiguous bf16 -> one 16B store
                union { bf16_t e[8]; uint4 q; } t;
                #pragma unroll
                for (int i = 0; i < 8; ++i) {
                    float v = a[i] * g.alpha + bv;
                    if (g.relu) v = fmaxf(v, 0.f);
                    t.e[i] = (bf16_t)v;
                }
                *(uint4*)(Cbp + (size_t)col * g.ldc + rowb + lh * 8) = t.q;
            } else {
                #pragma unroll
                for (int i = 0; i < 8; ++i) {
                    int row = rowb + lh * 8 + i;      // C layout: VGPR i -> M = i + 8*half
                    float v = a[i] * g.alpha + bv;
                    if (g.relu) v = fmaxf(v, 0.f);
                    if (Cfp) Cfp[(size_t)row * g.ldc + col] = v;
                    if (Cbp) Cbp[(size_t)row * g.ldc + col] = (bf16_t)v;
                }
            }
        }
    }
}

// ---------- elementwise / helper kernels ----------
__global__ void k_cast_f32_bf16(const float* s, bf16_t* d, long n) {
    long i = (long)blockIdx.x * blockDim.x + threadIdx.x;
    long st = (long)gridDim.x * blockDim.x;
    for (; i < n; i += st) d[i] = (bf16_t)s[i];
}

// cast + transpose: src (R x C) f32 -> dst (C x R) bf16
__global__ void k_cast_t(const float* s, bf16_t* d, int R, int C) {
    long i = (long)blockIdx.x * blockDim.x + threadIdx.x;
    if (i >= (long)R * C) return;
    int r = (int)(i / C), c = (int)(i % C);
    d[(size_t)c * R + r] = (bf16_t)s[i];
}

__global__ void k_gather_emb(const int* idx, const float* emb, bf16_t* out) {
    int bn = blockIdx.x, t = threadIdx.x;                 // 256 threads = E
    out[(size_t)bn * 256 + t] = (bf16_t)emb[(size_t)idx[bn] * 256 + t];
}

// Wff1 (256x400) -> Wff1t (416x256), zero pad rows (pad output cols)
__global__ void k_pad_wff1t(const float* W, bf16_t* Wd) {
    long i = (long)blockIdx.x * blockDim.x + threadIdx.x;
    if (i >= 416L * 256) return;
    int n = (int)(i / 256), k = (int)(i % 256);
    Wd[i] = (n < 400) ? (bf16_t)W[k * 400 + n] : (bf16_t)0.f;
}
__global__ void k_pad_bff1(const float* b, float* bd) {
    int i = blockIdx.x * blockDim.x + threadIdx.x;
    if (i < 416) bd[i] = (i < 400) ? b[i] : 0.f;
}
// Wff2 (400x256) -> Wff2t (256x416), zero pad K columns
__global__ void k_pad_wff2t(const float* W, bf16_t* Wd) {
    long i = (long)blockIdx.x * blockDim.x + threadIdx.x;
    if (i >= 256L * 416) return;
    int n = (int)(i / 416), k = (int)(i % 416);
    Wd[i] = (k < 400) ? (bf16_t)W[(size_t)k * 256 + n] : (bf16_t)0.f;
}

// in-place row softmax on bf16 scores; one block per row
__global__ __launch_bounds__(256) void k_softmax(bf16_t* S, int L) {
    __shared__ float red[256];
    size_t base = (size_t)blockIdx.x * L;
    int t = threadIdx.x;
    float m = -1e30f;
    for (int i = t; i < L; i += 256) m = fmaxf(m, (float)S[base + i]);
    red[t] = m; __syncthreads();
    for (int s = 128; s > 0; s >>= 1) { if (t < s) red[t] = fmaxf(red[t], red[t + s]); __syncthreads(); }
    float mx = red[0]; __syncthreads();
    float sum = 0.f;
    for (int i = t; i < L; i += 256) sum += __expf((float)S[base + i] - mx);
    red[t] = sum; __syncthreads();
    for (int s = 128; s > 0; s >>= 1) { if (t < s) red[t] += red[t + s]; __syncthreads(); }
    float inv = 1.f / red[0];
    for (int i = t; i < L; i += 256) S[base + i] = (bf16_t)(__expf((float)S[base + i] - mx) * inv);
}

// y = LN(x + d) * g + b ; one 256-thread block per row (H == 256)
__global__ __launch_bounds__(256)
void k_res_ln(const float* X, int xRows, int xOff, const float* D,
              const float* gamma, const float* beta,
              float* Of, bf16_t* Ob, int S) {
    __shared__ float red[256];
    int t = threadIdx.x, blk = blockIdx.x;
    int b = blk / S, s = blk % S;
    float v = X[((size_t)b * xRows + xOff + s) * 256 + t] + D[(size_t)blk * 256 + t];
    red[t] = v; __syncthreads();
    for (int k = 128; k > 0; k >>= 1) { if (t < k) red[t] += red[t + k]; __syncthreads(); }
    float mu = red[0] * (1.f / 256.f); __syncthreads();
    float d = v - mu;
    red[t] = d * d; __syncthreads();
    for (int k = 128; k > 0; k >>= 1) { if (t < k) red[t] += red[t + k]; __syncthreads(); }
    float y = d * rsqrtf(red[0] * (1.f / 256.f) + 1e-5f) * gamma[t] + beta[t];
    if (Of) Of[(size_t)blk * 256 + t] = y;
    if (Ob) Ob[(size_t)blk * 256 + t] = (bf16_t)y;
}

__global__ void k_mean(const float* X, float* Out, int S) {
    int b = blockIdx.x, t = threadIdx.x;
    float acc = 0.f;
    for (int s = 0; s < S; ++s) acc += X[((size_t)b * S + s) * 256 + t];
    Out[b * 256 + t] = acc / (float)S;
}

// m = p - n + f  (affine fc collapses: fc(p)-fc(n)+fc(f) = (p-n+f)@W + b)
__global__ void k_combine(const float* M0, const float* M1, const float* M2, bf16_t* Out) {
    int i = blockIdx.x * 256 + threadIdx.x;
    Out[i] = (bf16_t)(M0[i] - M1[i] + M2[i]);
}

// ---------- host orchestration ----------
extern "C" void kernel_launch(void* const* d_in, const int* in_sizes, int n_in,
                              void* d_out, int out_size, void* d_ws, size_t ws_size,
                              hipStream_t stream) {
    const int*   hyper = (const int*)  d_in[0];
    const float* HT    = (const float*)d_in[1];
    const float* emb   = (const float*)d_in[2];
    const float* Wg1   = (const float*)d_in[3];  const float* bg1  = (const float*)d_in[4];
    const float* Wg2   = (const float*)d_in[5];  const float* bg2  = (const float*)d_in[6];
    const float* Wqkv  = (const float*)d_in[7];  const float* bqkv = (const float*)d_in[8];
    const float* Wo    = (const float*)d_in[9];  const float* bo   = (const float*)d_in[10];
    const float* ln1g  = (const float*)d_in[11]; const float* ln1b = (const float*)d_in[12];
    const float* Wff1  = (const float*)d_in[13]; const float* bff1 = (const float*)d_in[14];
    const float* Wff2  = (const float*)d_in[15]; const float* bff2 = (const float*)d_in[16];
    const float* ln2g  = (const float*)d_in[17]; const float* ln2b = (const float*)d_in[18];
    const float* fcW   = (const float*)d_in[19]; const float* fcb  = (const float*)d_in[20];
    float* out = (float*)d_out;

    const int B = 16, N = 1024, H = 256;

    // workspace carve-up
    char* w = (char*)d_ws;
    auto alloc = [&](size_t bytes) { void* p = w; w += (bytes + 255) & ~(size_t)255; return p; };
    // Sb (attention scores, bf16, 16*4*512*512) aliases HTb (bf16 HT) -- graph phase done first
    bf16_t* HTb  = (bf16_t*)alloc(33554432);  bf16_t* Sb = HTb;
    bf16_t* Ab   = (bf16_t*)alloc((size_t)B*N*H*2);      // gathered / hidden bf16
    bf16_t* Tt   = (bf16_t*)alloc((size_t)B*N*H*2);      // (h @ Wg)^T : (b, 256, 1024)
    float*  Hf   = (float*) alloc((size_t)B*N*H*4);      // encoder input, f32
    bf16_t* Hb   = (bf16_t*)alloc((size_t)B*N*H*2);
    bf16_t* QKVb = (bf16_t*)alloc((size_t)B*512*512*2);  // q,k only: (b, S, 512)
    bf16_t* Vt   = (bf16_t*)alloc((size_t)B*256*512*2);  // v^T: (b, 256, S)
    bf16_t* CTXb = (bf16_t*)alloc((size_t)B*512*256*2);
    float*  Df   = (float*) alloc((size_t)B*512*256*4);
    float*  Yf   = (float*) alloc((size_t)B*512*256*4);
    bf16_t* Yb   = (bf16_t*)alloc((size_t)B*512*256*2);
    bf16_t* FFb  = (bf16_t*)alloc((size_t)B*512*416*2);
    float*  X2f  = (float*) alloc((size_t)B*512*256*4);
    float*  Mean = (float*) alloc((size_t)3*16*256*4);
    bf16_t* Mb   = (bf16_t*)alloc((size_t)16*256*2);
    bf16_t* Wg1t = (bf16_t*)alloc(65536*2);
    bf16_t* Wg2t = (bf16_t*)alloc(65536*2);
    bf16_t* Wqkvt= (bf16_t*)alloc(196608*2);             // (768, 256)
    bf16_t* Wot  = (bf16_t*)alloc(65536*2);
    bf16_t* Wff1t= (bf16_t*)alloc((size_t)416*256*2);
    bf16_t* Wff2t= (bf16_t*)alloc((size_t)256*416*2);
    bf16_t* fcWt = (bf16_t*)alloc(65536*2);
    float*  bff1p= (float*) alloc(416*4);

    auto gemm = [&](const bf16_t* A, int lda, long long sAb, long long sAh,
                    const bf16_t* Bm, int ldb, long long sBb, long long sBh,
                    float* Cf, bf16_t* Cb, int ldc, long long sCb, long long sCh, int transC,
                    const float* bias, int M, int Nn, int K, int NB, int NH,
                    float alpha, int relu) {
        GemmArgs g{A, Bm, bias, Cf, Cb, M, Nn, K, lda, ldb, ldc,
                   sAb, sAh, sBb, sBh, sCb, sCh, NH, transC, relu, alpha};
        dim3 grid((Nn + 127) / 128, (M + 63) / 64, NB * NH);
        dim3 blk(32, 8);
        k_gemm_bf16<<<grid, blk, 0, stream>>>(g);
    };

    // ---- precision casts / transposes ----
    k_cast_f32_bf16<<<4096, 256, 0, stream>>>(HT, HTb, (long)B * N * N);
    k_cast_t<<<256, 256, 0, stream>>>(Wg1,  Wg1t, 256, 256);
    k_cast_t<<<256, 256, 0, stream>>>(Wg2,  Wg2t, 256, 256);
    k_cast_t<<<768, 256, 0, stream>>>(Wqkv, Wqkvt, 256, 768);
    k_cast_t<<<256, 256, 0, stream>>>(Wo,   Wot,  256, 256);
    k_cast_t<<<256, 256, 0, stream>>>(fcW,  fcWt, 256, 256);
    k_pad_wff1t<<<(416 * 256 + 255) / 256, 256, 0, stream>>>(Wff1, Wff1t);
    k_pad_bff1<<<2, 256, 0, stream>>>(bff1, bff1p);
    k_pad_wff2t<<<(256 * 416 + 255) / 256, 256, 0, stream>>>(Wff2, Wff2t);
    k_gather_emb<<<B * N, 256, 0, stream>>>(hyper, emb, Ab);

    // ---- graph layers: h = HT @ (h @ Wg) + bg ----
    const long long sBN = (long long)N * H;          // 1024*256
    // Tt = (Ab @ Wg1)^T  (transC store)
    gemm(Ab, H, sBN, 0, Wg1t, H, 0, 0, nullptr, Tt, N, sBN, 0, 1, nullptr, N, H, H, B, 1, 1.f, 0);
    // Ab = HTb @ Tt^T + bg1
    gemm(HTb, N, (long long)N * N, 0, Tt, N, sBN, 0, nullptr, Ab, H, sBN, 0, 0, bg1, N, H, N, B, 1, 1.f, 0);
    gemm(Ab, H, sBN, 0, Wg2t, H, 0, 0, nullptr, Tt, N, sBN, 0, 1, nullptr, N, H, H, B, 1, 1.f, 0);
    gemm(HTb, N, (long long)N * N, 0, Tt, N, sBN, 0, Hf, Hb, H, sBN, 0, 0, bg2, N, H, N, B, 1, 1.f, 0);

    // ---- encoder on 3 slices ----
    const int offs[3] = {0, 512, 768};
    const int lens[3] = {512, 256, 256};
    for (int seg = 0; seg < 3; ++seg) {
        const int off = offs[seg], S = lens[seg];
        // q,k = x @ Wqkv[:, :512] + bqkv[:512]
        gemm(Hb + (size_t)off * H, H, sBN, 0, Wqkvt, H, 0, 0,
             nullptr, QKVb, 512, (long long)S * 512, 0, 0, bqkv, S, 512, H, B, 1, 1.f, 0);
        // v^T = (x @ Wqkv[:, 512:] + bqkv[512:])^T  -> Vt (b, 256, S)
        gemm(Hb + (size_t)off * H, H, sBN, 0, Wqkvt + (size_t)512 * H, H, 0, 0,
             nullptr, Vt, S, (long long)256 * S, 0, 1, bqkv + 512, S, 256, H, B, 1, 1.f, 0);
        // scores = (q @ k^T) / 8, per (batch, head); k rows are K-contiguous
        gemm(QKVb, 512, (long long)S * 512, 64, QKVb + 256, 512, (long long)S * 512, 64,
             nullptr, Sb, S, (long long)4 * S * S, (long long)S * S, 0, nullptr,
             S, S, 64, B, 4, 0.125f, 0);
        k_softmax<<<B * 4 * S, 256, 0, stream>>>(Sb, S);
        // ctx = probs @ v ; B = Vt rows (head h -> rows 64h..64h+63), K-contiguous
        gemm(Sb, S, (long long)4 * S * S, (long long)S * S, Vt, S, (long long)256 * S, (long long)64 * S,
             nullptr, CTXb, 256, (long long)S * 256, 64, 0, nullptr, S, 64, S, B, 4, 1.f, 0);
        // proj = ctx @ Wo + bo
        gemm(CTXb, H, (long long)S * H, 0, Wot, H, 0, 0,
             Df, nullptr, H, (long long)S * H, 0, 0, bo, S, H, H, B, 1, 1.f, 0);
        // x1 = LN(x + proj)
        k_res_ln<<<B * S, 256, 0, stream>>>(Hf, N, off, Df, ln1g, ln1b, Yf, Yb, S);
        // ff = relu(x1 @ Wff1 + bff1)   (400 padded to 416 with zeros)
        gemm(Yb, H, (long long)S * H, 0, Wff1t, H, 0, 0,
             nullptr, FFb, 416, (long long)S * 416, 0, 0, bff1p, S, 416, H, B, 1, 1.f, 1);
        // ff2 = ff @ Wff2 + bff2        (zero K-pad contributes 0)
        gemm(FFb, 416, (long long)S * 416, 0, Wff2t, 416, 0, 0,
             Df, nullptr, H, (long long)S * H, 0, 0, bff2, S, H, 416, B, 1, 1.f, 0);
        // x2 = LN(x1 + ff2)
        k_res_ln<<<B * S, 256, 0, stream>>>(Yf, S, 0, Df, ln2g, ln2b, X2f, nullptr, S);
        k_mean<<<B, 256, 0, stream>>>(X2f, Mean + seg * B * H, S);
    }

    // ---- fc(p)-fc(n)+fc(f) = (p-n+f)@W + b ; then relu ----
    k_combine<<<16, 256, 0, stream>>>(Mean, Mean + 4096, Mean + 8192, Mb);
    gemm(Mb, H, 0, 0, fcWt, H, 0, 0, out, nullptr, H, 0, 0, 0, fcb, 16, H, H, 1, 1, 1.f, 1);
}
